// Network_29119878267194
// MI455X (gfx1250) — matrix operand order repaired and checked by reading, not compiled, not run on hardware
//
#include <hip/hip_runtime.h>
#include <hip/hip_bf16.h>
#include <cmath>

typedef _Float16 f16t;
typedef __attribute__((ext_vector_type(16))) _Float16 v16h;
typedef __attribute__((ext_vector_type(8)))  float    v8f;

static constexpr int T_   = 256;
static constexpr int B_   = 32;
static constexpr int N_   = 128;
static constexpr int H_   = 128;
static constexpr int M_   = B_ * N_;   // 4096 independent LSTM rows
static constexpr int G_   = 4 * H_;    // 512 gate columns
static constexpr int ROWS = 16;        // rows per workgroup (one WMMA M-tile)
static constexpr int HSTR = 136;       // padded f16 stride (bank-conflict free)
static constexpr int WSTR = 136;

__device__ __forceinline__ float sigm(float x)     { return 1.0f / (1.0f + __expf(-x)); }
__device__ __forceinline__ float softplus(float z) { return log1pf(__expf(z)); }

__global__ __launch_bounds__(256)
void lstm_neuron_kernel(const float* __restrict__ input, const float* __restrict__ W_ih,
                        const float* __restrict__ b_ih,  const float* __restrict__ W_hh,
                        const float* __restrict__ b_hh,  const float* __restrict__ Wl,
                        const float* __restrict__ bl,    const float* __restrict__ W1,
                        const float* __restrict__ b1,    const float* __restrict__ W2,
                        const float* __restrict__ b2,    float* __restrict__ out)
{
    extern __shared__ char smem[];
    f16t*  whh_s = (f16t*)smem;                  // [512][WSTR] f16 : W_hh (row = gate col)
    f16t*  h_s   = whh_s + G_ * WSTR;            // [2][16][HSTR] f16 : double-buffered h
    float* xrow  = (float*)(h_s + 2*ROWS*HSTR);  // [2][16] : double-buffered x_t
    float* outp  = xrow + 2*ROWS;                // [2][16] : double-buffered out_{t-1}

    const int tid  = threadIdx.x;
    const int row0 = blockIdx.x * ROWS;

    // WMMA lane mapping (wave32). Wave wv owns hidden units j = wv*16 + mrow and
    // computes all four gates for them: gate g lives at column tile g*128 + wv*16.
    const int L    = tid & 31;
    const int kh   = L >> 4;                     // lane half
    const int mrow = L & 15;
    const int wv   = tid >> 5;                   // wave id 0..7
    const int jcol = wv * 16 + mrow;             // hidden unit owned per lane

    // gat/output thread mapping: row r, 16 lanes per row (aligned half-wave)
    const int r   = tid >> 4;
    const int l16 = tid & 15;

    // ---------------- one-time setup ----------------
    for (int i = tid; i < G_ * H_; i += 256) {   // W_hh f32 -> f16 LDS, padded rows
        int col = i >> 7, k = i & (H_ - 1);
        whh_s[col * WSTR + k] = (f16t)W_hh[i];
    }
    for (int i = tid; i < 2 * ROWS * HSTR; i += 256) h_s[i] = (f16t)0.0f;   // h0 = 0

    // Hoisted per-lane constants (all register-resident, zero per-step LDS reads):
    float w0c[4], w1c[4], cbc[4];                // rank-2 x2 @ W_ih^T + combined bias
    #pragma unroll
    for (int g = 0; g < 4; ++g) {
        int col = g * 128 + jcol;
        w0c[g] = W_ih[col * 2 + 0];
        w1c[g] = W_ih[col * 2 + 1];
        cbc[g] = b_ih[col] + b_hh[col];
    }
    float wlr[3][8];                             // Wl slice for the gat dot products
    #pragma unroll
    for (int g = 0; g < 3; ++g)
        #pragma unroll
        for (int jj = 0; jj < 8; ++jj) wlr[g][jj] = Wl[g * 128 + l16 * 8 + jj];
    const float bl0 = bl[0], bl1 = bl[1], bl2 = bl[2];

    // neuron MLP collapses to affine: out_lin = An*x + Bn
    const int nn = (row0 + r) & (N_ - 1);
    float An = 0.f, Bn = 0.f;
    for (int k = 0; k < 10; ++k) { An += W1[nn*10+k] * W2[nn*10+k]; Bn += b1[nn*10+k] * W2[nn*10+k]; }
    Bn += b2[nn];

    if (l16 == 0) {                              // t = 0 output + prefetch x_1
        float x0 = input[row0 + r];
        float o0 = softplus(An * x0 + Bn - 1.0f);
        out[row0 + r] = o0;
        outp[1 * ROWS + r] = o0;
        xrow[1 * ROWS + r] = input[M_ + row0 + r];
    }

    float creg[8];                               // c-state lives in registers
    #pragma unroll
    for (int v = 0; v < 8; ++v) creg[v] = 0.f;

    for (int t = 1; t < T_; ++t) {
        const int rb = t & 1, wb = rb ^ 1;       // read/write buffer indices
        __syncthreads();                          // single barrier per step

        // prefetch x_{t+1} into the write buffer (hidden behind the WMMAs)
        if (l16 == 0 && t + 1 < T_) xrow[wb * ROWS + r] = input[(t + 1) * M_ + row0 + r];

        // ---- WMMA: acc[g][v] = (h_{t-1} @ W_hh^T)[row = v+8*kh][col = g*128 + jcol]
        v8f acc[4];
        #pragma unroll
        for (int g = 0; g < 4; ++g)
            #pragma unroll
            for (int v = 0; v < 8; ++v) acc[g][v] = 0.f;

        const f16t* hb = h_s + rb * ROWS * HSTR;
        #pragma unroll
        for (int kc = 0; kc < 4; ++kc) {
            // A fragment: lane holds K = kc*32 + kh*16 + 0..15
            v16h a;
            const f16t* hp = hb + mrow * HSTR + kc * 32 + kh * 16;
            #pragma unroll
            for (int i = 0; i < 8; ++i) a[i]     = hp[i];
            #pragma unroll
            for (int i = 0; i < 8; ++i) a[8 + i] = hp[8 + i];

            #pragma unroll
            for (int g = 0; g < 4; ++g) {
                // B fragment: lane (n=mrow) holds K = kc*32 + kh*16 + 0..15 of column
                const f16t* wp = whh_s + (g * 128 + wv * 16 + mrow) * WSTR + kc * 32 + kh * 16;
                v16h bm;
                #pragma unroll
                for (int i = 0; i < 16; ++i) bm[i] = wp[i];
                acc[g] = __builtin_amdgcn_wmma_f32_16x16x32_f16(
                    false, a, false, bm, (short)0, acc[g], false, false);
            }
        }

        // ---- gat(h_{t-1}) + output (h_{t-1} is the read buffer this step)
        float s0 = 0.f, s1 = 0.f, s2 = 0.f;
        {
            const f16t* hq = hb + r * HSTR + l16 * 8;   // one b128: 8 f16, conflict-free
            #pragma unroll
            for (int jj = 0; jj < 8; ++jj) {
                float hv = (float)hq[jj];
                s0 += hv * wlr[0][jj];
                s1 += hv * wlr[1][jj];
                s2 += hv * wlr[2][jj];
            }
            #pragma unroll
            for (int d = 8; d >= 1; d >>= 1) {           // 16-lane half-wave reduce
                s0 += __shfl_xor(s0, d, 32);
                s1 += __shfl_xor(s1, d, 32);
                s2 += __shfl_xor(s2, d, 32);
            }
        }
        if (l16 == 0) {
            float gg, aa, th;
            if (t == 1) { gg = 1.f; aa = 1.f; th = 1.f; }          // carry0 = ones
            else        { gg = s0 + bl0; aa = s1 + bl1; th = s2 + bl2; }
            float x  = xrow[rb * ROWS + r];
            float ol = An * x + Bn;
            float o  = gg * softplus(aa * (ol - th)) / aa;
            out[t * M_ + row0 + r] = o;
            outp[wb * ROWS + r] = o;                               // out_prev for t+1
        }

        // ---- finish gates in registers: rank-2 x2 term + bias, LSTM cell, write h_t
        f16t* hw = h_s + wb * ROWS * HSTR;
        #pragma unroll
        for (int v = 0; v < 8; ++v) {
            int row  = v + 8 * kh;
            float x  = xrow[rb * ROWS + row];     // broadcast LDS reads
            float op = outp[rb * ROWS + row];
            float gi = acc[0][v] + x * w0c[0] + op * w1c[0] + cbc[0];
            float gf = acc[1][v] + x * w0c[1] + op * w1c[1] + cbc[1];
            float gG = acc[2][v] + x * w0c[2] + op * w1c[2] + cbc[2];
            float go = acc[3][v] + x * w0c[3] + op * w1c[3] + cbc[3];
            float c  = sigm(gf) * creg[v] + sigm(gi) * tanhf(gG);
            creg[v]  = c;
            float h  = sigm(go) * tanhf(c);
            hw[row * HSTR + jcol] = (f16t)h;      // only h round-trips LDS
        }
    }
}

extern "C" void kernel_launch(void* const* d_in, const int* in_sizes, int n_in,
                              void* d_out, int out_size, void* d_ws, size_t ws_size,
                              hipStream_t stream) {
    const float* input = (const float*)d_in[0];
    const float* W_ih  = (const float*)d_in[1];
    const float* b_ih  = (const float*)d_in[2];
    const float* W_hh  = (const float*)d_in[3];
    const float* b_hh  = (const float*)d_in[4];
    const float* Wl    = (const float*)d_in[5];
    const float* bl    = (const float*)d_in[6];
    const float* W1    = (const float*)d_in[7];
    const float* b1    = (const float*)d_in[8];
    const float* W2    = (const float*)d_in[9];
    const float* b2    = (const float*)d_in[10];
    float* out = (float*)d_out;

    size_t shmem = (size_t)(G_ * WSTR + 2 * ROWS * HSTR) * sizeof(f16t)
                 + (size_t)(4 * ROWS) * sizeof(float);

    (void)hipFuncSetAttribute(reinterpret_cast<const void*>(&lstm_neuron_kernel),
                              hipFuncAttributeMaxDynamicSharedMemorySize, (int)shmem);

    lstm_neuron_kernel<<<M_ / ROWS, 256, shmem, stream>>>(
        input, W_ih, b_ih, W_hh, b_hh, Wl, bl, W1, b1, W2, b2, out);
}
